// RWKV_Tmix_x070_v2_36910948942407
// MI455X (gfx1250) — matrix operand order repaired and checked by reading, not compiled, not run on hardware
//
#include <hip/hip_runtime.h>
#include <hip/hip_bf16.h>
#include <math.h>

// ---------------------------------------------------------------------------
// RWKV-7 Tmix for MI455X (gfx1250): bf16 WMMA GEMMs (compile-time K, ping-pong
// fragment pipeline, 64x32 strip per wave) + fused elementwise + chunked LDS
// double-buffered scan. wave32 everywhere.
// ---------------------------------------------------------------------------

typedef __bf16 bf16_t;
typedef __attribute__((ext_vector_type(16))) __bf16 v16bf;
typedef __attribute__((ext_vector_type(8)))  __bf16 v8bf;
typedef __attribute__((ext_vector_type(8)))  float   v8f;

#define EP_NONE    0
#define EP_TANH    1
#define EP_SIGMOID 2
#define EP_LORA_W  3   // -softplus(-(bias+acc)) - 0.5

// ---------------------------------------------------------------------------
// fp32 -> bf16 convert (weights)
// ---------------------------------------------------------------------------
__global__ __launch_bounds__(256)
void f2bf_kernel(const float* __restrict__ s, bf16_t* __restrict__ d, size_t n) {
  size_t i = (size_t)blockIdx.x * blockDim.x + threadIdx.x;
  if (i < n) d[i] = (bf16_t)s[i];
}

// fp32 (R,Cc) -> bf16 (Cc,R) transpose-convert (LoRA weights -> [N,K] layout)
__global__ __launch_bounds__(256)
void tr_f2bf_kernel(const float* __restrict__ s, bf16_t* __restrict__ d, int R, int Cc) {
  size_t i = (size_t)blockIdx.x * blockDim.x + threadIdx.x;
  size_t n = (size_t)R * Cc;
  if (i < n) {
    int cc = (int)(i % Cc);
    int r  = (int)(i / Cc);
    d[(size_t)cc * R + r] = (bf16_t)s[i];
  }
}

// ---------------------------------------------------------------------------
// Token shift + six lerps, written as bf16 GEMM inputs
// ---------------------------------------------------------------------------
__global__ __launch_bounds__(256)
void mix_kernel(const float* __restrict__ x,
                const float* __restrict__ mr, const float* __restrict__ mw,
                const float* __restrict__ mk, const float* __restrict__ mv,
                const float* __restrict__ ma, const float* __restrict__ mg,
                bf16_t* __restrict__ xr, bf16_t* __restrict__ xw,
                bf16_t* __restrict__ xk, bf16_t* __restrict__ xv,
                bf16_t* __restrict__ xa, bf16_t* __restrict__ xg,
                int T, int C, size_t total) {
  size_t i = (size_t)blockIdx.x * blockDim.x + threadIdx.x;
  if (i >= total) return;
  int    c  = (int)(i % C);
  size_t bt = i / C;
  int    t  = (int)(bt % T);
  float  x0 = x[i];
  float  xp = (t == 0) ? 0.f : x[i - C];
  float  xx = xp - x0;
  xr[i] = (bf16_t)(x0 + xx * mr[c]);
  xw[i] = (bf16_t)(x0 + xx * mw[c]);
  xk[i] = (bf16_t)(x0 + xx * mk[c]);
  xv[i] = (bf16_t)(x0 + xx * mv[c]);
  xa[i] = (bf16_t)(x0 + xx * ma[c]);
  xg[i] = (bf16_t)(x0 + xx * mg[c]);
}

// ---------------------------------------------------------------------------
// Generic bf16 WMMA GEMM: C[M,N] = epilogue(bias[n] + A[M,K] * Bt[N,K]^T)
// K is a compile-time constant -> the ping-pong indices (s&1) constant-fold
// under unrolling: steady-state loop is loads + v_wmma + staggered waits with
// no copies and no scalar branches. One wave owns a (16*MT)x(16*NT) strip.
// K multiple of 32; N multiple of 16*NT; M multiple of 16*MT.
// ---------------------------------------------------------------------------
template <int MT, int NT, int K>
__global__ __launch_bounds__(256)
void gemm_bf16_wmma(const bf16_t* A, const bf16_t* Bt,
                    float* Cf, bf16_t* Cb, const float* bias,
                    int M, int N, int epilogue) {
  const int wid    = blockIdx.x * (blockDim.x >> 5) + (threadIdx.x >> 5);
  const int tilesN = N / (16 * NT);
  const int total  = (M / (16 * MT)) * tilesN;
  if (wid >= total) return;                 // wave-uniform: EXEC stays all-1s
  const int tn   = wid % tilesN;
  const int tm   = wid / tilesN;
  const int m0   = tm * 16 * MT;
  const int n0   = tn * 16 * NT;
  const int lane = threadIdx.x & 31;
  const int l    = lane & 15;
  const int half = lane >> 4;

  v8f acc[MT][NT];
#pragma unroll
  for (int i = 0; i < MT; ++i)
#pragma unroll
    for (int n = 0; n < NT; ++n)
#pragma unroll
      for (int j = 0; j < 8; ++j) acc[i][n][j] = 0.f;

  // Fragment base pointers (this code's contiguous k order, same for A and B):
  // A 16x32: lane l holds row m0+l; half selects K {0..15} vs {16..31} contiguous
  // B 32x16: lane l holds col n0+l; half selects K {0..15} vs {16..31} contiguous
  const bf16_t* aptr = A  + (size_t)(m0 + l) * K + half * 16;
  const bf16_t* bptr = Bt + (size_t)(n0 + l) * K + half * 16;

  constexpr int NSTEPS = K / 32;
  v8bf  alo[2][MT], ahi[2][MT];
  v16bf bfr[2][NT];

#pragma unroll
  for (int nn = 0; nn < NT; ++nn)
    bfr[0][nn] = *(const v16bf*)(bptr + (size_t)nn * 16 * K);
#pragma unroll
  for (int ii = 0; ii < MT; ++ii) {
    alo[0][ii] = *(const v8bf*)(aptr + (size_t)ii * 16 * K);
    ahi[0][ii] = *(const v8bf*)(aptr + (size_t)ii * 16 * K + 8);
  }

#pragma unroll 2
  for (int s = 0; s < NSTEPS - 1; ++s) {
    const int cur = s & 1;
    const int nxt = cur ^ 1;
    const int kn  = (s + 1) * 32;
    if (s + 2 < NSTEPS) {
      __builtin_prefetch(bptr + kn + 32, 0, 1);   // global_prefetch_b8
      __builtin_prefetch(aptr + kn + 32, 0, 1);
    }
#pragma unroll
    for (int nn = 0; nn < NT; ++nn)
      bfr[nxt][nn] = *(const v16bf*)(bptr + (size_t)nn * 16 * K + kn);
#pragma unroll
    for (int ii = 0; ii < MT; ++ii) {
      alo[nxt][ii] = *(const v8bf*)(aptr + (size_t)ii * 16 * K + kn);
      ahi[nxt][ii] = *(const v8bf*)(aptr + (size_t)ii * 16 * K + kn + 8);
    }
#pragma unroll
    for (int ii = 0; ii < MT; ++ii) {
      v16bf afrag = __builtin_shufflevector(alo[cur][ii], ahi[cur][ii],
          0, 1, 2, 3, 4, 5, 6, 7, 8, 9, 10, 11, 12, 13, 14, 15);
#pragma unroll
      for (int nn = 0; nn < NT; ++nn)
        acc[ii][nn] = __builtin_amdgcn_wmma_f32_16x16x32_bf16(
            false, afrag, false, bfr[cur][nn], (short)0, acc[ii][nn], false, false);
    }
  }
  {
    const int cur = (NSTEPS - 1) & 1;
#pragma unroll
    for (int ii = 0; ii < MT; ++ii) {
      v16bf afrag = __builtin_shufflevector(alo[cur][ii], ahi[cur][ii],
          0, 1, 2, 3, 4, 5, 6, 7, 8, 9, 10, 11, 12, 13, 14, 15);
#pragma unroll
      for (int nn = 0; nn < NT; ++nn)
        acc[ii][nn] = __builtin_amdgcn_wmma_f32_16x16x32_bf16(
            false, afrag, false, bfr[cur][nn], (short)0, acc[ii][nn], false, false);
    }
  }

#pragma unroll
  for (int nn = 0; nn < NT; ++nn) {
    const int   col = n0 + nn * 16 + l;
    const float bv  = bias ? bias[col] : 0.f;
#pragma unroll
    for (int ii = 0; ii < MT; ++ii) {
#pragma unroll
      for (int vv = 0; vv < 8; ++vv) {
        float val = acc[ii][nn][vv] + bv;
        if      (epilogue == EP_TANH)    val = tanhf(val);
        else if (epilogue == EP_SIGMOID) val = 1.f / (1.f + __expf(-val));
        else if (epilogue == EP_LORA_W)  val = -log1pf(__expf(-val)) - 0.5f;
        size_t row = (size_t)(m0 + ii * 16 + half * 8 + vv);
        if (Cf) Cf[row * N + col] = val;
        else    Cb[row * N + col] = (bf16_t)val;
      }
    }
  }
}

// ---------------------------------------------------------------------------
// Per-token prep: kk L2-norm per head, k/v/decay/aa/bb for the scan.
// One block per (b,t); 16 lanes per head; shfl reductions within wave32.
// k/v/d written over kraw/vraw/w in place (same index, no restrict).
// ---------------------------------------------------------------------------
__global__ __launch_bounds__(256)
void prep_kernel(const float* kraw, const float* a, const float* w,
                 const float* vraw, const float* vmix, const float* vfirst,
                 const float* __restrict__ k_k, const float* __restrict__ k_a,
                 float* kout, float* dout, float* vout,
                 float* __restrict__ aaout, float* __restrict__ bbout, int C) {
  size_t base = (size_t)blockIdx.x * C;
  int    tid  = threadIdx.x;
  int    c0   = tid * 4;
  float  kk[4], kr[4], av[4], ss = 0.f;
#pragma unroll
  for (int j = 0; j < 4; ++j) {
    int c = c0 + j;
    kr[j] = kraw[base + c];
    av[j] = a[base + c];
    kk[j] = kr[j] * k_k[c];
    ss += kk[j] * kk[j];
  }
  ss += __shfl_xor(ss, 1); ss += __shfl_xor(ss, 2);
  ss += __shfl_xor(ss, 4); ss += __shfl_xor(ss, 8);
  float inv = 1.f / fmaxf(sqrtf(ss), 1e-12f);
#pragma unroll
  for (int j = 0; j < 4; ++j) {
    int   c   = c0 + j;
    float kkn = kk[j] * inv;
    kout[base + c] = kr[j] * (1.f + (av[j] - 1.f) * k_a[c]);
    dout[base + c] = __expf(-__expf(w[base + c]));
    float vr = vraw[base + c];
    vout[base + c] = vr + vmix[base + c] * (vfirst[base + c] - vr);
    aaout[base + c] = -kkn;
    bbout[base + c] = kkn * av[j];
  }
}

// ---------------------------------------------------------------------------
// Sequential RWKV-7 scan, chunked + LDS double-buffered.
// One block per (b,h); 64x64 fp32 state in VGPRs: thread (i=tid>>2, jq=tid&3)
// owns S[i][jq*16..jq*16+15]. While computing a 16-step chunk from one LDS
// buffer, the next chunk's vectors are in flight (one float4 global load per
// array per thread), drained into the other buffer with one barrier per chunk.
// ---------------------------------------------------------------------------
#define SCAN_CH 16
__global__ __launch_bounds__(256)
void scan_kernel(const float* __restrict__ r, const float* __restrict__ d,
                 const float* __restrict__ k, const float* __restrict__ v,
                 const float* __restrict__ aa, const float* __restrict__ bb,
                 float* __restrict__ y, int T, int H) {
  const int N = 64;
  const int C = H * N;
  int bh  = blockIdx.x;
  int b   = bh / H, h = bh % H;
  int tid = threadIdx.x;
  int i   = tid >> 2;
  int jq  = tid & 3;
  __shared__ float buf[2][6][SCAN_CH * 64];
  float S[16];
#pragma unroll
  for (int j = 0; j < 16; ++j) S[j] = 0.f;

  size_t base = (size_t)b * T * C + (size_t)h * N;
  const float* srcs[6] = {r, d, k, v, aa, bb};

  // loader slot: timestep lt = tid/16, channels lj..lj+3 = (tid%16)*4
  int lt = tid >> 4;
  int lj = (tid & 15) << 2;

  {
    size_t off = base + (size_t)lt * C + lj;
#pragma unroll
    for (int s = 0; s < 6; ++s)
      *(float4*)&buf[0][s][lt * 64 + lj] = *(const float4*)(srcs[s] + off);
  }
  __syncthreads();

  const int nch = T / SCAN_CH;
  for (int c = 0; c < nch; ++c) {
    const int  pp   = c & 1;
    const bool more = (c + 1 < nch);
    float4 rg[6];
    if (more) {
      size_t off = base + ((size_t)(c + 1) * SCAN_CH + lt) * C + lj;
#pragma unroll
      for (int s = 0; s < 6; ++s) rg[s] = *(const float4*)(srcs[s] + off);
    }
    const float* lr = buf[pp][0];
    const float* ld = buf[pp][1];
    const float* lk = buf[pp][2];
    const float* lv = buf[pp][3];
    const float* la = buf[pp][4];
    const float* lb = buf[pp][5];
    for (int tt = 0; tt < SCAN_CH; ++tt) {
      const int o = tt * 64;
      float sa = 0.f;
#pragma unroll
      for (int j = 0; j < 16; ++j) sa += S[j] * la[o + jq * 16 + j];
      sa += __shfl_xor(sa, 1); sa += __shfl_xor(sa, 2);
      float vi = lv[o + i];
      float yv = 0.f;
#pragma unroll
      for (int j = 0; j < 16; ++j) {
        int   jc = o + jq * 16 + j;
        float s  = S[j] * ld[jc] + sa * lb[jc] + vi * lk[jc];
        S[j] = s;
        yv += s * lr[jc];
      }
      yv += __shfl_xor(yv, 1); yv += __shfl_xor(yv, 2);
      if (jq == 0) y[base + (size_t)(c * SCAN_CH + tt) * C + i] = yv;
    }
    if (more) {
#pragma unroll
      for (int s = 0; s < 6; ++s)
        *(float4*)&buf[pp ^ 1][s][lt * 64 + lj] = rg[s];
    }
    __syncthreads();
  }
}

// ---------------------------------------------------------------------------
// GroupNorm(H) + r*k*r_k residual + gate -> bf16 input of the output GEMM.
// One block per (b,t); 16 lanes per head; shfl reductions within wave32.
// ---------------------------------------------------------------------------
__global__ __launch_bounds__(256)
void post_kernel(const float* __restrict__ y, const float* __restrict__ r,
                 const float* __restrict__ k, const float* __restrict__ v,
                 const float* __restrict__ g, const float* __restrict__ r_k,
                 const float* __restrict__ ln_w, const float* __restrict__ ln_b,
                 bf16_t* __restrict__ yg, int C) {
  size_t base = (size_t)blockIdx.x * C;
  int    tid  = threadIdx.x;
  int    c0   = tid * 4;
  float  yv[4], s1 = 0.f, s2 = 0.f, rk = 0.f;
#pragma unroll
  for (int j = 0; j < 4; ++j) {
    int c = c0 + j;
    yv[j] = y[base + c];
    s1 += yv[j];
    s2 += yv[j] * yv[j];
    rk += r[base + c] * k[base + c] * r_k[c];
  }
  s1 += __shfl_xor(s1, 1); s1 += __shfl_xor(s1, 2);
  s1 += __shfl_xor(s1, 4); s1 += __shfl_xor(s1, 8);
  s2 += __shfl_xor(s2, 1); s2 += __shfl_xor(s2, 2);
  s2 += __shfl_xor(s2, 4); s2 += __shfl_xor(s2, 8);
  rk += __shfl_xor(rk, 1); rk += __shfl_xor(rk, 2);
  rk += __shfl_xor(rk, 4); rk += __shfl_xor(rk, 8);
  float mu  = s1 * (1.f / 64.f);
  float var = s2 * (1.f / 64.f) - mu * mu;
  float inv = rsqrtf(var + 64e-5f);
#pragma unroll
  for (int j = 0; j < 4; ++j) {
    int   c = c0 + j;
    float o = (yv[j] - mu) * inv * ln_w[c] + ln_b[c] + rk * v[base + c];
    yg[base + c] = (bf16_t)(o * g[base + c]);
  }
}

// ---------------------------------------------------------------------------
// Launcher
// ---------------------------------------------------------------------------
template <int K>
static inline void launch_gemm(const bf16_t* A, const bf16_t* Bt, float* Cf, bf16_t* Cb,
                               const float* bias, int M, int N, int ep,
                               hipStream_t st) {
  int waves  = (M / 64) * (N / 32);
  int blocks = (waves + 7) / 8;
  gemm_bf16_wmma<4, 2, K><<<blocks, 256, 0, st>>>(A, Bt, Cf, Cb, bias, M, N, ep);
}

extern "C" void kernel_launch(void* const* d_in, const int* in_sizes, int n_in,
                              void* d_out, int out_size, void* d_ws, size_t ws_size,
                              hipStream_t stream) {
  (void)in_sizes; (void)n_in; (void)out_size; (void)ws_size;
  const int B = 2, T = 4096, C = 1024, H = 16;
  const size_t BT = (size_t)B * T, BTC = BT * C;

  const float* x      = (const float*)d_in[0];
  const float* vfirst = (const float*)d_in[1];
  const float* x_r = (const float*)d_in[2], *x_w = (const float*)d_in[3];
  const float* x_k = (const float*)d_in[4], *x_v = (const float*)d_in[5];
  const float* x_a = (const float*)d_in[6], *x_g = (const float*)d_in[7];
  const float* w0 = (const float*)d_in[8],  *w1 = (const float*)d_in[9],  *w2 = (const float*)d_in[10];
  const float* a0 = (const float*)d_in[11], *a1 = (const float*)d_in[12], *a2 = (const float*)d_in[13];
  const float* v0 = (const float*)d_in[14], *v1 = (const float*)d_in[15], *v2 = (const float*)d_in[16];
  const float* g1 = (const float*)d_in[17], *g2 = (const float*)d_in[18];
  const float* k_k = (const float*)d_in[19], *k_a = (const float*)d_in[20];
  const float* r_k = (const float*)d_in[21];
  const float* Wr = (const float*)d_in[22], *Wk = (const float*)d_in[23];
  const float* Wv = (const float*)d_in[24], *Wo = (const float*)d_in[25];
  const float* ln_w = (const float*)d_in[26], *ln_b = (const float*)d_in[27];

  // ---- carve workspace (256B aligned) ----
  unsigned char* wsp = (unsigned char*)d_ws;
  auto carve = [&](size_t bytes) -> void* {
    void* p = (void*)wsp;
    wsp += (bytes + 255) & ~(size_t)255;
    return p;
  };
  // fp32 intermediates
  float* rbuf    = (float*)carve(BTC * 4);
  float* kbuf    = (float*)carve(BTC * 4);  // kraw -> k (in place in prep)
  float* vbuf    = (float*)carve(BTC * 4);  // vraw -> v
  float* wbuf    = (float*)carve(BTC * 4);  // w    -> decay
  float* abuf    = (float*)carve(BTC * 4);
  float* vmixbuf = (float*)carve(BTC * 4);
  float* gbuf    = (float*)carve(BTC * 4);
  float* aabuf   = (float*)carve(BTC * 4);
  float* bbbuf   = (float*)carve(BTC * 4);
  float* ybuf    = (float*)carve(BTC * 4);
  // bf16 GEMM operands
  bf16_t* xr = (bf16_t*)carve(BTC * 2); bf16_t* xw = (bf16_t*)carve(BTC * 2);
  bf16_t* xk = (bf16_t*)carve(BTC * 2); bf16_t* xv = (bf16_t*)carve(BTC * 2);
  bf16_t* xa = (bf16_t*)carve(BTC * 2); bf16_t* xg = (bf16_t*)carve(BTC * 2);
  bf16_t* WrB = (bf16_t*)carve((size_t)C * C * 2);
  bf16_t* WkB = (bf16_t*)carve((size_t)C * C * 2);
  bf16_t* WvB = (bf16_t*)carve((size_t)C * C * 2);
  bf16_t* WoB = (bf16_t*)carve((size_t)C * C * 2);
  bf16_t* w1T = (bf16_t*)carve((size_t)C * 64 * 2);
  bf16_t* w2T = (bf16_t*)carve((size_t)C * 64 * 2);
  bf16_t* a1T = (bf16_t*)carve((size_t)C * 64 * 2);
  bf16_t* a2T = (bf16_t*)carve((size_t)C * 64 * 2);
  bf16_t* v1T = (bf16_t*)carve((size_t)C * 32 * 2);
  bf16_t* v2T = (bf16_t*)carve((size_t)C * 32 * 2);
  bf16_t* g1T = (bf16_t*)carve((size_t)C * 160 * 2);
  bf16_t* g2T = (bf16_t*)carve((size_t)C * 160 * 2);
  bf16_t* hw  = (bf16_t*)carve(BT * 64 * 2);
  bf16_t* ha  = (bf16_t*)carve(BT * 64 * 2);
  bf16_t* hv  = (bf16_t*)carve(BT * 32 * 2);
  bf16_t* hg  = (bf16_t*)carve(BT * 160 * 2);
  bf16_t* yg  = (bf16_t*)carve(BTC * 2);

  auto gridFor = [](size_t n) { return (unsigned)((n + 255) / 256); };

  // ---- 1) weight conversion (L2-resident bf16 copies) ----
  size_t nCC = (size_t)C * C;
  f2bf_kernel<<<gridFor(nCC), 256, 0, stream>>>(Wr, WrB, nCC);
  f2bf_kernel<<<gridFor(nCC), 256, 0, stream>>>(Wk, WkB, nCC);
  f2bf_kernel<<<gridFor(nCC), 256, 0, stream>>>(Wv, WvB, nCC);
  f2bf_kernel<<<gridFor(nCC), 256, 0, stream>>>(Wo, WoB, nCC);
  tr_f2bf_kernel<<<gridFor((size_t)C * 64), 256, 0, stream>>>(w1, w1T, C, 64);
  tr_f2bf_kernel<<<gridFor((size_t)C * 64), 256, 0, stream>>>(w2, w2T, 64, C);
  tr_f2bf_kernel<<<gridFor((size_t)C * 64), 256, 0, stream>>>(a1, a1T, C, 64);
  tr_f2bf_kernel<<<gridFor((size_t)C * 64), 256, 0, stream>>>(a2, a2T, 64, C);
  tr_f2bf_kernel<<<gridFor((size_t)C * 32), 256, 0, stream>>>(v1, v1T, C, 32);
  tr_f2bf_kernel<<<gridFor((size_t)C * 32), 256, 0, stream>>>(v2, v2T, 32, C);
  tr_f2bf_kernel<<<gridFor((size_t)C * 160), 256, 0, stream>>>(g1, g1T, C, 160);
  tr_f2bf_kernel<<<gridFor((size_t)C * 160), 256, 0, stream>>>(g2, g2T, 160, C);

  // ---- 2) token shift + lerps ----
  mix_kernel<<<gridFor(BTC), 256, 0, stream>>>(x, x_r, x_w, x_k, x_v, x_a, x_g,
                                               xr, xw, xk, xv, xa, xg, T, C, BTC);

  // ---- 3) GEMMs (all v_wmma_f32_16x16x32_bf16) ----
  launch_gemm<1024>(xr, WrB, rbuf, nullptr, nullptr, (int)BT, C, EP_NONE, stream);
  launch_gemm<1024>(xk, WkB, kbuf, nullptr, nullptr, (int)BT, C, EP_NONE, stream);
  launch_gemm<1024>(xv, WvB, vbuf, nullptr, nullptr, (int)BT, C, EP_NONE, stream);
  launch_gemm<1024>(xw, w1T, nullptr, hw, nullptr, (int)BT, 64, EP_TANH, stream);
  launch_gemm<1024>(xa, a1T, nullptr, ha, nullptr, (int)BT, 64, EP_NONE, stream);
  launch_gemm<1024>(xv, v1T, nullptr, hv, nullptr, (int)BT, 32, EP_NONE, stream);
  launch_gemm<1024>(xg, g1T, nullptr, hg, nullptr, (int)BT, 160, EP_SIGMOID, stream);
  launch_gemm<64>(hw, w2T, wbuf, nullptr, w0, (int)BT, C, EP_LORA_W, stream);
  launch_gemm<64>(ha, a2T, abuf, nullptr, a0, (int)BT, C, EP_SIGMOID, stream);
  launch_gemm<32>(hv, v2T, vmixbuf, nullptr, v0, (int)BT, C, EP_SIGMOID, stream);
  launch_gemm<160>(hg, g2T, gbuf, nullptr, nullptr, (int)BT, C, EP_NONE, stream);

  // ---- 4) per-token prep for the scan ----
  prep_kernel<<<(unsigned)BT, 256, 0, stream>>>(kbuf, abuf, wbuf, vbuf, vmixbuf, vfirst,
                                                k_k, k_a, kbuf, wbuf, vbuf, aabuf, bbbuf, C);

  // ---- 5) sequential RWKV-7 scan (B*H = 32 chains, LDS double-buffered) ----
  scan_kernel<<<B * H, 256, 0, stream>>>(rbuf, wbuf, kbuf, vbuf, aabuf, bbbuf, ybuf, T, H);

  // ---- 6) groupnorm + residual + gate ----
  post_kernel<<<(unsigned)BT, 256, 0, stream>>>(ybuf, rbuf, kbuf, vbuf, gbuf,
                                                r_k, ln_w, ln_b, yg, C);

  // ---- 7) output projection -> d_out (fp32) ----
  launch_gemm<1024>(yg, WoB, (float*)d_out, nullptr, nullptr, (int)BT, C, EP_NONE, stream);
}
